// C2f_CA_DCN_79070347919453
// MI455X (gfx1250) — matrix operand order repaired and checked by reading, not compiled, not run on hardware
//
#include <hip/hip_runtime.h>
#include <math.h>

typedef __bf16 v16bf __attribute__((ext_vector_type(16)));
typedef __bf16 v2bf  __attribute__((ext_vector_type(2)));
typedef float  v8f   __attribute__((ext_vector_type(8)));

#define NPIX 16384   // B*H*W = 4*64*64

// native bf16 converts -> v_cvt_pk_bf16_f32 on gfx1250
__device__ __forceinline__ unsigned short f2bf(float f) {
  __bf16 h = (__bf16)f;
  return __builtin_bit_cast(unsigned short, h);
}
__device__ __forceinline__ unsigned int pk2(float a, float b) {
  v2bf p;
  p.x = (__bf16)a;
  p.y = (__bf16)b;
  return __builtin_bit_cast(unsigned int, p);
}

union Frag { uint4 q[2]; v16bf v; };

// ---------------------------------------------------------------------------
// Fused WMMA GEMM: C[M,N] = act( (A[M,K] @ W[K,N] + bias) * gamma + beta )
// Tile: BM=128, BN=64, BK=64. 256 threads = 8 waves laid out 4(M) x 2(N),
// each wave owns a 32x32 patch -> 8 x v_wmma_f32_16x16x32_bf16 per K-tile.
// Software pipelined: LDS store -> barrier -> issue next global loads ->
// 8 WMMAs -> barrier, so HBM latency overlaps matrix math.
// act: 0=none 1=silu 2=relu 3=sigmoid. bias/gamma/beta may be null.
// ---------------------------------------------------------------------------
__global__ __launch_bounds__(256)
void wmma_gemm_kernel(const float* __restrict__ A, int lda,
                      const float* __restrict__ W, int ldb,
                      float* __restrict__ C, int ldc, int K,
                      const float* __restrict__ bias,
                      const float* __restrict__ gamma,
                      const float* __restrict__ beta,
                      int act)
{
  __shared__ unsigned short As[128][72];  // [m][k] bf16, 144B stride: 16B-aligned frags
  __shared__ unsigned short Bs[64][72];   // [n][k] bf16 (transposed weight tile)

  const int tid  = threadIdx.x;
  const int m0   = blockIdx.y * 128;
  const int n0   = blockIdx.x * 64;
  const int wave = tid >> 5, lane = tid & 31;
  const int wm   = wave >> 1, wn = wave & 1;

  const v8f zero = {0.f,0.f,0.f,0.f,0.f,0.f,0.f,0.f};
  v8f acc[2][2];
  acc[0][0] = zero; acc[0][1] = zero; acc[1][0] = zero; acc[1][1] = zero;

  // global->LDS assignments
  const int arow = tid >> 1;          // 0..127
  const int acol = (tid & 1) << 5;    // 0 or 32 (floats)
  const int bk   = tid >> 2;          // 0..63
  const int bn   = (tid & 3) << 4;    // 0,16,32,48

  const float* ap = A + (size_t)(m0 + arow) * lda + acol;
  const float* wp = W + (size_t)bk * ldb + n0 + bn;

  // pipeline prologue: first tile into registers
  float4 ar[8], br[4];
  #pragma unroll
  for (int i = 0; i < 8; ++i) ar[i] = *(const float4*)(ap + 4 * i);
  #pragma unroll
  for (int i = 0; i < 4; ++i) br[i] = *(const float4*)(wp + 4 * i);

  // fragment addressing (A and B fragments: 16 contiguous K per lane half,
  // lanes 0-15 K{0..15}, lanes 16-31 K{16..31})
  const int rlo = lane & 15;
  const int kcA = (lane >> 4) << 4;   // 0 or 16
  const int kcB = (lane >> 4) << 4;   // 0 or 16
  const int ra0 = wm * 32 + rlo, ra1 = ra0 + 16;
  const int cb0 = wn * 32 + rlo, cb1 = cb0 + 16;

  for (int k0 = 0; k0 < K; k0 += 64) {
    // ---- convert + store current tile to LDS (v_cvt_pk_bf16_f32 path)
    #pragma unroll
    for (int i = 0; i < 4; ++i) {
      uint4 q;
      q.x = pk2(ar[2 * i].x,     ar[2 * i].y);
      q.y = pk2(ar[2 * i].z,     ar[2 * i].w);
      q.z = pk2(ar[2 * i + 1].x, ar[2 * i + 1].y);
      q.w = pk2(ar[2 * i + 1].z, ar[2 * i + 1].w);
      *(uint4*)&As[arow][acol + 8 * i] = q;
    }
    #pragma unroll
    for (int i = 0; i < 4; ++i) {
      Bs[bn + 4 * i + 0][bk] = f2bf(br[i].x);
      Bs[bn + 4 * i + 1][bk] = f2bf(br[i].y);
      Bs[bn + 4 * i + 2][bk] = f2bf(br[i].z);
      Bs[bn + 4 * i + 3][bk] = f2bf(br[i].w);
    }
    __syncthreads();

    // ---- issue next tile's global loads; waits land at next LDS store
    ap += 64;
    wp += (size_t)64 * ldb;
    if (k0 + 64 < K) {
      #pragma unroll
      for (int i = 0; i < 8; ++i) ar[i] = *(const float4*)(ap + 4 * i);
      #pragma unroll
      for (int i = 0; i < 4; ++i) br[i] = *(const float4*)(wp + 4 * i);
    }

    // ---- 2 x (4 WMMA) from LDS
    #pragma unroll
    for (int kk = 0; kk < 64; kk += 32) {
      Frag fa0, fa1, fb0, fb1;
      fa0.q[0] = *(const uint4*)&As[ra0][kk + kcA];
      fa0.q[1] = *(const uint4*)&As[ra0][kk + kcA + 8];
      fa1.q[0] = *(const uint4*)&As[ra1][kk + kcA];
      fa1.q[1] = *(const uint4*)&As[ra1][kk + kcA + 8];
      fb0.q[0] = *(const uint4*)&Bs[cb0][kk + kcB];
      fb0.q[1] = *(const uint4*)&Bs[cb0][kk + kcB + 8];
      fb1.q[0] = *(const uint4*)&Bs[cb1][kk + kcB];
      fb1.q[1] = *(const uint4*)&Bs[cb1][kk + kcB + 8];

      acc[0][0] = __builtin_amdgcn_wmma_f32_16x16x32_bf16(false, fa0.v, false, fb0.v,
                                                          (short)0, acc[0][0], false, false);
      acc[0][1] = __builtin_amdgcn_wmma_f32_16x16x32_bf16(false, fa0.v, false, fb1.v,
                                                          (short)0, acc[0][1], false, false);
      acc[1][0] = __builtin_amdgcn_wmma_f32_16x16x32_bf16(false, fa1.v, false, fb0.v,
                                                          (short)0, acc[1][0], false, false);
      acc[1][1] = __builtin_amdgcn_wmma_f32_16x16x32_bf16(false, fa1.v, false, fb1.v,
                                                          (short)0, acc[1][1], false, false);
    }
    __syncthreads();
  }

  // ---- epilogue + store (C/D layout: lanes 0-15 rows j, lanes 16-31 rows j+8)
  const int rhi = (lane >> 4) << 3;
  #pragma unroll
  for (int sn = 0; sn < 2; ++sn) {
    const int col = n0 + wn * 32 + sn * 16 + rlo;
    const float bi = bias  ? bias[col]  : 0.f;
    const float ga = gamma ? gamma[col] : 1.f;
    const float be = beta  ? beta[col]  : 0.f;
    #pragma unroll
    for (int sm = 0; sm < 2; ++sm) {
      const int rb = m0 + wm * 32 + sm * 16 + rhi;
      #pragma unroll
      for (int j = 0; j < 8; ++j) {
        float v = acc[sm][sn][j] + bi;
        v = v * ga + be;
        if (act == 1)      v = v / (1.f + __expf(-v));
        else if (act == 2) v = fmaxf(v, 0.f);
        else if (act == 3) v = 1.f / (1.f + __expf(-v));
        C[(size_t)(rb + j) * ldc + col] = v;
      }
    }
  }
}

// ---------------------------------------------------------------------------
// Depthwise 3x3 (SAME, zero pad) + bias + LayerNorm(C) + exact GELU.
// One wave per pixel; lane owns channels c = lane + 32*j (coalesced).
// ---------------------------------------------------------------------------
__global__ __launch_bounds__(256)
void dw_ln_gelu_kernel(const float* __restrict__ x, const float* __restrict__ kern,
                       const float* __restrict__ kb, const float* __restrict__ lg,
                       const float* __restrict__ lb, float* __restrict__ out)
{
  const int wave = threadIdx.x >> 5, lane = threadIdx.x & 31;
  const int pix = blockIdx.x * 8 + wave;
  const int b = pix >> 12, hw = pix & 4095, h = hw >> 6, w = hw & 63;

  float v[8];
  #pragma unroll
  for (int j = 0; j < 8; ++j) {
    const int c = lane + 32 * j;
    float accv = kb[c];
    #pragma unroll
    for (int kh = 0; kh < 3; ++kh) {
      const int hh = h + kh - 1;
      if ((unsigned)hh >= 64u) continue;
      #pragma unroll
      for (int kw = 0; kw < 3; ++kw) {
        const int ww = w + kw - 1;
        if ((unsigned)ww >= 64u) continue;
        accv += x[(((size_t)(b * 64 + hh) * 64) + ww) * 256 + c] *
                kern[(kh * 3 + kw) * 256 + c];
      }
    }
    v[j] = accv;
  }
  float s = 0.f, s2 = 0.f;
  #pragma unroll
  for (int j = 0; j < 8; ++j) { s += v[j]; s2 += v[j] * v[j]; }
  for (int o = 16; o; o >>= 1) { s += __shfl_xor(s, o); s2 += __shfl_xor(s2, o); }
  const float mean = s * (1.f / 256.f);
  const float var  = s2 * (1.f / 256.f) - mean * mean;
  const float rstd = rsqrtf(var + 1e-5f);
  #pragma unroll
  for (int j = 0; j < 8; ++j) {
    const int c = lane + 32 * j;
    float t = (v[j] - mean) * rstd * lg[c] + lb[c];
    out[(size_t)pix * 256 + c] = 0.5f * t * (1.f + erff(t * 0.70710678118f));
  }
}

// ---------------------------------------------------------------------------
// offsets (18) + mask logits (9) projections, K=256, then softmax over 9.
// One thread per pixel; activation tile + weights staged in LDS.
// om layout per pixel: [0..17]=offset, [18..26]=softmaxed mask.
// ---------------------------------------------------------------------------
__global__ __launch_bounds__(256)
void offset_mask_kernel(const float* __restrict__ t1, const float* __restrict__ Wo,
                        const float* __restrict__ bo, const float* __restrict__ Wm,
                        const float* __restrict__ bm, float* __restrict__ om)
{
  __shared__ float xs[256][33];
  __shared__ float ws[32][28];
  const int tid  = threadIdx.x;
  const int pix0 = blockIdx.x * 256;

  float acc[27];
  #pragma unroll
  for (int i = 0; i < 27; ++i) acc[i] = 0.f;

  for (int kc = 0; kc < 256; kc += 32) {
    __syncthreads();
    #pragma unroll
    for (int p = 0; p < 8; ++p) {                 // coalesced 128B runs
      const int r  = p * 32 + (tid >> 3);
      const int c4 = (tid & 7) * 4;
      float4 f = *(const float4*)(t1 + (size_t)(pix0 + r) * 256 + kc + c4);
      xs[r][c4 + 0] = f.x; xs[r][c4 + 1] = f.y;
      xs[r][c4 + 2] = f.z; xs[r][c4 + 3] = f.w;
    }
    for (int i = tid; i < 32 * 27; i += 256) {
      const int k = i / 27, j = i % 27;
      ws[k][j] = (j < 18) ? Wo[(size_t)(kc + k) * 18 + j]
                          : Wm[(size_t)(kc + k) * 9 + (j - 18)];
    }
    __syncthreads();
    #pragma unroll 4
    for (int k = 0; k < 32; ++k) {
      const float xv = xs[tid][k];
      #pragma unroll
      for (int j = 0; j < 27; ++j) acc[j] += xv * ws[k][j];
    }
  }

  const size_t base = (size_t)(pix0 + tid) * 27;
  #pragma unroll
  for (int j = 0; j < 18; ++j) om[base + j] = acc[j] + bo[j];
  float e[9], mx = -1e30f;
  #pragma unroll
  for (int j = 0; j < 9; ++j) { e[j] = acc[18 + j] + bm[j]; mx = fmaxf(mx, e[j]); }
  float ssum = 0.f;
  #pragma unroll
  for (int j = 0; j < 9; ++j) { e[j] = __expf(e[j] - mx); ssum += e[j]; }
  const float inv = 1.f / ssum;
  #pragma unroll
  for (int j = 0; j < 9; ++j) om[base + 18 + j] = e[j] * inv;
}

// ---------------------------------------------------------------------------
// DCNv3 core: y[pix,c] = sum_p mask_p * bilinear(xproj, ref+grid+off)_c
// One wave per pixel; lane owns channels {4*lane..+3} and {128+4*lane..+3}.
// ---------------------------------------------------------------------------
__global__ __launch_bounds__(256)
void dcn_core_kernel(const float* __restrict__ xproj, const float* __restrict__ om,
                     float* __restrict__ y)
{
  const int wave = threadIdx.x >> 5, lane = threadIdx.x & 31;
  const int pix = blockIdx.x * 8 + wave;
  const int b = pix >> 12, hw = pix & 4095, h = hw >> 6, w = hw & 63;
  const float* omp = om + (size_t)pix * 27;
  const int cA = lane * 4, cB = 128 + lane * 4;

  float4 aA = {0.f,0.f,0.f,0.f}, aB = {0.f,0.f,0.f,0.f};

  #pragma unroll
  for (int p = 0; p < 9; ++p) {
    const int gi = p / 3, gj = p % 3;            // x-major flatten: dx slow, dy fast
    const float ox = omp[2 * p], oy = omp[2 * p + 1];
    const float m  = omp[18 + p];
    const float pxu = (float)(w + gi - 1) + ox;  // unpadded sample coords
    const float pyu = (float)(h + gj - 1) + oy;
    const float fx = floorf(pxu), fy = floorf(pyu);
    const int x0 = (int)fx, y0 = (int)fy;
    const float wx1 = pxu - fx, wy1 = pyu - fy;
    const float wx0 = 1.f - wx1, wy0 = 1.f - wy1;

    const int   xc[4] = { x0, x0 + 1, x0,     x0 + 1 };
    const int   yc[4] = { y0, y0,     y0 + 1, y0 + 1 };
    const float wc[4] = { wy0 * wx0 * m, wy0 * wx1 * m, wy1 * wx0 * m, wy1 * wx1 * m };
    #pragma unroll
    for (int q = 0; q < 4; ++q) {
      if ((unsigned)xc[q] < 64u && (unsigned)yc[q] < 64u) {
        const float* src = xproj + (((size_t)(b * 64 + yc[q]) * 64 + xc[q]) << 8);
        const float wt = wc[q];
        float4 va = *(const float4*)(src + cA);
        float4 vb = *(const float4*)(src + cB);
        aA.x = fmaf(wt, va.x, aA.x); aA.y = fmaf(wt, va.y, aA.y);
        aA.z = fmaf(wt, va.z, aA.z); aA.w = fmaf(wt, va.w, aA.w);
        aB.x = fmaf(wt, vb.x, aB.x); aB.y = fmaf(wt, vb.y, aB.y);
        aB.z = fmaf(wt, vb.z, aB.z); aB.w = fmaf(wt, vb.w, aB.w);
      }
    }
  }
  float* dst = y + (size_t)pix * 256;
  *(float4*)(dst + cA) = aA;
  *(float4*)(dst + cB) = aB;
}

// ---------------------------------------------------------------------------
// Coordinate attention: pooling, apply, concat copy
// ---------------------------------------------------------------------------
__global__ __launch_bounds__(256)
void ca_pool_kernel(const float* __restrict__ y, float* __restrict__ tcat)
{
  const int idx = blockIdx.x * 256 + threadIdx.x;   // over B*128*256
  const int c = idx & 255, r = (idx >> 8) & 127, b = idx >> 15;
  float s = 0.f;
  if (r < 64) {           // pooled over width, row indexed by h
    const float* p = y + ((size_t)(b * 64 + r) * 64) * 256 + c;
    for (int w = 0; w < 64; ++w) s += p[(size_t)w * 256];
  } else {                // pooled over height, row indexed by w
    const float* p = y + ((size_t)b * 4096 + (r - 64)) * 256 + c;
    for (int h = 0; h < 64; ++h) s += p[(size_t)h * 64 * 256];
  }
  tcat[idx] = s * (1.f / 64.f);
}

__global__ __launch_bounds__(256)
void ca_apply_kernel(const float* __restrict__ y, const float* __restrict__ xa,
                     const float* __restrict__ ya, float* __restrict__ outc)
{
  const int pix = blockIdx.x, c = threadIdx.x;
  const int b = pix >> 12, hw = pix & 4095, h = hw >> 6, w = hw & 63;
  const float v = y[(size_t)pix * 256 + c]
                * xa[(size_t)(b * 128 + w) * 256 + c]          // s[:, :W] @ W2x rows
                * ya[(size_t)(b * 128 + 64 + h) * 256 + c];    // s[:, W:] @ W2y rows
  outc[(size_t)pix * 768 + c] = v;
}

__global__ __launch_bounds__(256)
void concat_copy_kernel(const float* __restrict__ src, float* __restrict__ dst)
{
  const int pix = blockIdx.x, c = threadIdx.x;
  dst[(size_t)pix * 768 + c] = src[(size_t)pix * 256 + c];
}

// ---------------------------------------------------------------------------
// Host orchestration
// ---------------------------------------------------------------------------
namespace {
struct Dcn {
  const float *b1,*b2,*cw,*Wi,*Wm,*Wo,*Wout,*bi,*bm,*bo,*bout,*dwb,*dwk,*lnb,*lng,*g1,*g2;
};
struct Ca { const float *W1,*W2x,*W2y,*b1,*b2x,*b2y,*bb,*g; };
}

extern "C" void kernel_launch(void* const* d_in, const int* in_sizes, int n_in,
                              void* d_out, int out_size, void* d_ws, size_t ws_size,
                              hipStream_t stream)
{
  (void)in_sizes; (void)n_in; (void)out_size; (void)ws_size;
  auto F = [&](int i) { return (const float*)d_in[i]; };
  const float* x_in = F(0);

  // pytree flatten order: dict keys sorted at each level, lists in order.
  auto getDcn = [&](int b) {
    Dcn d;
    d.b1 = F(b + 0);  d.b2 = F(b + 1);  d.cw = F(b + 2);
    d.Wi = F(b + 3);  d.Wm = F(b + 4);  d.Wo = F(b + 5);  d.Wout = F(b + 6);
    d.bi = F(b + 7);  d.bm = F(b + 8);  d.bo = F(b + 9);  d.bout = F(b + 10);
    d.dwb = F(b + 11); d.dwk = F(b + 12); d.lnb = F(b + 13); d.lng = F(b + 14);
    d.g1 = F(b + 15); d.g2 = F(b + 16);
    return d;
  };
  auto getCa = [&](int b) {
    Ca c;
    c.W1 = F(b + 0); c.W2x = F(b + 1); c.W2y = F(b + 2); c.b1 = F(b + 3);
    c.b2x = F(b + 4); c.b2y = F(b + 5); c.bb = F(b + 6); c.g = F(b + 7);
    return c;
  };

  Ca  ca0  = getCa(1),   ca1  = getCa(43);
  Dcn b0d1 = getDcn(9),  b0d2 = getDcn(26);
  Dcn b1d1 = getDcn(51), b1d2 = getDcn(68);
  const float* cc_b = F(85);
  const float* cc_g = F(86);
  const float* cc_w = F(87);
  Dcn fin  = getDcn(88);
  Dcn init = getDcn(105);

  // workspace layout (floats)
  float* ws = (float*)d_ws;
  const size_t SL = (size_t)NPIX * 256;
  float* sA   = ws;                 // t0 / core
  float* sB   = ws + SL;            // xproj
  float* sC   = ws + 2 * SL;        // t1
  float* x0   = ws + 3 * SL;        // init output
  float* P    = ws + 4 * SL;
  float* Q    = ws + 5 * SL;
  float* cat  = ws + 6 * SL;                       // [NPIX, 768]
  float* om   = cat + (size_t)NPIX * 768;          // [NPIX, 27]
  float* tcat = om + (size_t)NPIX * 27;            // [512, 256]
  float* sbuf = tcat + (size_t)512 * 256;          // [512, 64]
  float* xab  = sbuf + (size_t)512 * 64;           // [512, 256]
  float* yab  = xab + (size_t)512 * 256;           // [512, 256]

  auto gemm = [&](const float* A, int lda, const float* W, int ldb, float* Cp, int ldc,
                  int M, int N, int K, const float* bias, const float* ga,
                  const float* be, int act) {
    dim3 grid(N / 64, M / 128);
    wmma_gemm_kernel<<<grid, dim3(256), 0, stream>>>(A, lda, W, ldb, Cp, ldc, K,
                                                     bias, ga, be, act);
  };

  auto run_dcn = [&](const float* in, int cin, const Dcn& p, float* out) {
    gemm(in, cin, p.cw, 256, sA, 256, NPIX, 256, cin, nullptr, p.g1, p.b1, 1); // SiLU(BN)
    gemm(sA, 256, p.Wi, 256, sB, 256, NPIX, 256, 256, p.bi, nullptr, nullptr, 0);
    dw_ln_gelu_kernel<<<NPIX / 8, 256, 0, stream>>>(sA, p.dwk, p.dwb, p.lng, p.lnb, sC);
    offset_mask_kernel<<<NPIX / 256, 256, 0, stream>>>(sC, p.Wo, p.bo, p.Wm, p.bm, om);
    dcn_core_kernel<<<NPIX / 8, 256, 0, stream>>>(sB, om, sA);   // sA reused for core
    gemm(sA, 256, p.Wout, 256, out, 256, NPIX, 256, 256, p.bout, p.g2, p.b2, 1);
  };

  auto run_ca = [&](const float* y, const Ca& c, float* catcol) {
    ca_pool_kernel<<<512, 256, 0, stream>>>(y, tcat);
    gemm(tcat, 256, c.W1, 64, sbuf, 64, 512, 64, 256, c.b1, c.g, c.bb, 2);   // ReLU(BN)
    gemm(sbuf, 64, c.W2x, 256, xab, 256, 512, 256, 64, c.b2x, nullptr, nullptr, 3);
    gemm(sbuf, 64, c.W2y, 256, yab, 256, 512, 256, 64, c.b2y, nullptr, nullptr, 3);
    ca_apply_kernel<<<NPIX, 256, 0, stream>>>(y, xab, yab, catcol);
  };

  // init
  run_dcn(x_in, 256, init, x0);
  // branch 0
  run_dcn(x0, 256, b0d1, P);
  run_dcn(P, 256, b0d2, Q);
  run_ca(Q, ca0, cat + 0);
  // branch 1
  run_dcn(x0, 256, b1d1, P);
  run_dcn(P, 256, b1d2, Q);
  run_ca(Q, ca1, cat + 256);
  // concat slot 2 = x0
  concat_copy_kernel<<<NPIX, 256, 0, stream>>>(x0, cat + 512);
  // cc conv 768->256 + BN + ReLU
  gemm(cat, 768, cc_w, 256, P, 256, NPIX, 256, 768, nullptr, cc_g, cc_b, 2);
  // final block writes straight to d_out
  run_dcn(P, 256, fin, (float*)d_out);
}
